// TransformerBlock_relative_3058016715287
// MI455X (gfx1250) — matrix operand order repaired and checked by reading, not compiled, not run on hardware
//
#include <hip/hip_runtime.h>

typedef __attribute__((ext_vector_type(16))) __bf16 v16bf;
typedef __attribute__((ext_vector_type(8)))  float  v8f;
typedef __attribute__((ext_vector_type(4)))  int    v4i;
typedef unsigned int   u32;
typedef unsigned short u16;

#if defined(__has_builtin)
#  if __has_builtin(__builtin_amdgcn_global_load_async_to_lds_b128)
#    define HAVE_ASYNC_LDS 1
#  endif
#endif
#ifndef HAVE_ASYNC_LDS
#  define HAVE_ASYNC_LDS 0
#endif

namespace {

constexpr int kB = 8, kS = 512, kH = 16, kDM = 1024, kDFF = 4096, kDK = 64;
constexpr int kM = kB * kS;   // 4096 token rows

static __device__ __forceinline__ u16 f2bf(float f) {
  u32 u = __float_as_uint(f);
  u32 r = u + 0x7FFFu + ((u >> 16) & 1u);   // round-to-nearest-even
  return (u16)(r >> 16);
}
static __device__ __forceinline__ float bf2f(u16 h) {
  return __uint_as_float(((u32)h) << 16);
}

struct __align__(16) U4 { u32 x, y, z, w; };
union AFrag { U4 q[2]; u32 u[8]; v16bf v; };

static __device__ __forceinline__ v8f wmma_bf(const AFrag& a, const AFrag& b, v8f c) {
  return __builtin_amdgcn_wmma_f32_16x16x32_bf16(false, a.v, false, b.v,
                                                 (short)0, c, false, false);
}

#if HAVE_ASYNC_LDS
// Async global->LDS 16-byte copy, tracked by ASYNCcnt (no VGPR round-trip).
// Builtin signature (from clang diagnostic): (v4i AS1*, v4i AS3*, imm, imm).
typedef __attribute__((address_space(1))) v4i* v4i_gptr;
typedef __attribute__((address_space(3))) v4i* v4i_lptr;
static __device__ __forceinline__ void async_ld_b128(const void* g, const void* l) {
  __builtin_amdgcn_global_load_async_to_lds_b128(
      (v4i_gptr)(unsigned long long)(size_t)g,
      (v4i_lptr)(u32)(size_t)l,
      0, 0);
}
static __device__ __forceinline__ void wait_async0() {
#if __has_builtin(__builtin_amdgcn_s_wait_asynccnt)
  __builtin_amdgcn_s_wait_asynccnt(0);
#else
  asm volatile("s_wait_asynccnt 0x0" ::: "memory");
#endif
}
#endif

// ---------------------------------------------------------------- fp32 -> bf16
__global__ void cvt_bf16_kernel(const float* __restrict__ src,
                                u16* __restrict__ dst, int n) {
  int i = blockIdx.x * blockDim.x + threadIdx.x;
  if (i < n) dst[i] = f2bf(src[i]);
}

// ---------------------------------------------------------------- LayerNorm
// torch semantics: std with ddof=1, y = g*(x-mean)/(std+eps)+b ; bf16 out
__global__ __launch_bounds__(256)
void ln_bf16_kernel(const float* __restrict__ x, const float* __restrict__ g,
                    const float* __restrict__ b, u16* __restrict__ out) {
  const int row = blockIdx.x;
  const float* xr = x + (size_t)row * kDM;
  float s = 0.f, s2 = 0.f;
  for (int i = threadIdx.x; i < kDM; i += 256) {
    float v = xr[i]; s += v; s2 += v * v;
  }
  __shared__ float red0[8], red1[8];
  for (int off = 16; off > 0; off >>= 1) {
    s  += __shfl_down(s,  off, 32);
    s2 += __shfl_down(s2, off, 32);
  }
  const int wid = threadIdx.x >> 5, lane = threadIdx.x & 31;
  if (lane == 0) { red0[wid] = s; red1[wid] = s2; }
  __syncthreads();
  if (wid == 0) {
    s  = (lane < 8) ? red0[lane] : 0.f;
    s2 = (lane < 8) ? red1[lane] : 0.f;
    for (int off = 4; off > 0; off >>= 1) {
      s  += __shfl_down(s,  off, 32);
      s2 += __shfl_down(s2, off, 32);
    }
    if (lane == 0) { red0[0] = s; red1[0] = s2; }
  }
  __syncthreads();
  const float mean = red0[0] / kDM;
  const float var  = (red1[0] - kDM * mean * mean) / (kDM - 1);
  const float inv  = 1.0f / (sqrtf(fmaxf(var, 0.f)) + 1e-6f);
  for (int i = threadIdx.x; i < kDM; i += 256) {
    float v = (xr[i] - mean) * inv * g[i] + b[i];
    out[(size_t)row * kDM + i] = f2bf(v);
  }
}

// ---------------------------------------------------------------- GEMM
// C[M,N] = A[M,K](bf16) * W[N,K]^T(bf16) + bias, epilogues:
//   EPI=0: bf16 out     EPI=1: GELU(tanh) -> bf16 out
//   EPI=2: + resid -> fp32 out
// Double-buffered LDS staging via GLOBAL_LOAD_ASYNC_TO_LDS_B128 (ASYNCcnt).
template<int EPI>
__global__ __launch_bounds__(256)
void gemm_bf16_kernel(const u16* __restrict__ A, const u16* __restrict__ W,
                      const float* __restrict__ bias, const float* __restrict__ resid,
                      u16* __restrict__ outb, float* __restrict__ outf,
                      int M, int N, int K) {
  constexpr int BM = 128, BN = 128, BK = 32, LST = 48;   // padded row (16B aligned)
  __shared__ __align__(16) u16 As[2][BM * LST];          // 2 x 12 KB
  __shared__ __align__(16) u16 Bs[2][BN * LST];          // 2 x 12 KB

  const int tid  = threadIdx.x;
  const int m0   = blockIdx.x * BM;
  const int n0   = blockIdx.y * BN;
  const int lane = tid & 31, w = tid >> 5;
  const int hf   = lane >> 4, l16 = lane & 15;
  const int wm   = w >> 1, wn = w & 1;

  const int ldRow = tid >> 1;          // 0..127
  const int ldCol = (tid & 1) * 16;    // 0 / 16

  v8f acc[2][4];
  v8f vzero = {};
#pragma unroll
  for (int i = 0; i < 2; ++i)
#pragma unroll
    for (int j = 0; j < 4; ++j) acc[i][j] = vzero;

  const u16* gA = A + (size_t)(m0 + ldRow) * K + ldCol;
  const u16* gW = W + (size_t)(n0 + ldRow) * K + ldCol;
  u16* lA = &As[0][ldRow * LST + ldCol];
  u16* lB = &Bs[0][ldRow * LST + ldCol];

  auto stage = [&](int buf, int k0) {
#if HAVE_ASYNC_LDS
    const int bo = buf * (BM * LST);
    async_ld_b128(gA + k0,     lA + bo);
    async_ld_b128(gA + k0 + 8, lA + bo + 8);
    async_ld_b128(gW + k0,     lB + bo);
    async_ld_b128(gW + k0 + 8, lB + bo + 8);
#else
    const int bo = buf * (BM * LST);
    U4 a0 = *(const U4*)(gA + k0);
    U4 a1 = *(const U4*)(gA + k0 + 8);
    U4 b0 = *(const U4*)(gW + k0);
    U4 b1 = *(const U4*)(gW + k0 + 8);
    *(U4*)(lA + bo)     = a0;
    *(U4*)(lA + bo + 8) = a1;
    *(U4*)(lB + bo)     = b0;
    *(U4*)(lB + bo + 8) = b1;
#endif
  };

  stage(0, 0);                             // prologue: fill buffer 0
  int buf = 0;
  for (int k0 = 0; k0 < K; k0 += BK, buf ^= 1) {
#if HAVE_ASYNC_LDS
    wait_async0();                         // this wave's async tiles landed
#endif
    __syncthreads();                       // all waves' tiles visible; prev buf free
    if (k0 + BK < K) {
      stage(buf ^ 1, k0 + BK);             // stream next slice into other buffer
      if (k0 + 2 * BK < K) {
        __builtin_prefetch(gA + k0 + 2 * BK, 0, 1);   // warm L2 two slices ahead
        __builtin_prefetch(gW + k0 + 2 * BK, 0, 1);
      }
    }

    // A fragment: dwords 0..3 = K[8h..8h+7], 4..7 = K[16+8h..23+8h]
    AFrag aF[2], bF[4];
#pragma unroll
    for (int tm = 0; tm < 2; ++tm) {
      const u16* base = &As[buf][(wm * 32 + tm * 16 + l16) * LST];
      aF[tm].q[0] = *(const U4*)(base + 8 * hf);
      aF[tm].q[1] = *(const U4*)(base + 16 + 8 * hf);
    }
    // B fragment: dwords 0..3 = K[8h..8h+7], 4..7 = K[16+8h..23+8h] for column l16
#pragma unroll
    for (int tn = 0; tn < 4; ++tn) {
      const u16* base = &Bs[buf][(wn * 64 + tn * 16 + l16) * LST];
      bF[tn].q[0] = *(const U4*)(base + 8 * hf);
      bF[tn].q[1] = *(const U4*)(base + 16 + 8 * hf);
    }
#pragma unroll
    for (int tm = 0; tm < 2; ++tm)
#pragma unroll
      for (int tn = 0; tn < 4; ++tn)
        acc[tm][tn] = wmma_bf(aF[tm], bF[tn], acc[tm][tn]);
  }

  // epilogue: C layout -> row = r + 8*hf, col = l16 within each 16x16 tile
#pragma unroll
  for (int tm = 0; tm < 2; ++tm)
#pragma unroll
    for (int tn = 0; tn < 4; ++tn) {
      const int col = n0 + wn * 64 + tn * 16 + l16;
      const float bcol = bias[col];
#pragma unroll
      for (int r = 0; r < 8; ++r) {
        const int row = m0 + wm * 32 + tm * 16 + r + 8 * hf;
        float c = acc[tm][tn][r] + bcol;
        if constexpr (EPI == 1) {
          float t = tanhf(0.7978845608028654f * (c + 0.044715f * c * c * c));
          c = 0.5f * c * (1.0f + t);
        }
        if constexpr (EPI == 2) {
          outf[(size_t)row * N + col] = c + resid[(size_t)row * N + col];
        } else {
          outb[(size_t)row * N + col] = f2bf(c);
        }
      }
    }
}

// ---------------------------------------------------------------- Attention
// Flash-style: one wave = one 16-row q tile of one (b,h). Q/K/V bf16 in
// [B*S, DM] with column = h*64+d. r_k/r_v stay fp32 (L2-resident, 128 MB).
__global__ __launch_bounds__(128)
void attn_kernel(const u16* __restrict__ Qg, const u16* __restrict__ Kg,
                 const u16* __restrict__ Vg, const float* __restrict__ rk,
                 const float* __restrict__ rvp, const int* __restrict__ mask,
                 u16* __restrict__ Og) {
  __shared__ __align__(16) float Qlds[4][16][kDK];  // 16 KB, fp32 Q for rel dot
  __shared__ __align__(16) float Pf[4][16][32];     // 8 KB, fp32 P for p@r_v
  __shared__ __align__(16) u16   Pb[4][16][32];     // 4 KB, bf16 P for WMMA

  const int tid  = threadIdx.x;
  const int w    = tid >> 5, lane = tid & 31;
  const int hf   = lane >> 4, l16 = lane & 15;

  const int blk  = blockIdx.x;
  const int bh   = blk >> 3;            // S/64 = 8 q-blocks per (b,h)
  const int qblk = blk & 7;
  const int b    = bh >> 4;
  const int h    = bh & 15;
  const int q0   = qblk * 64 + w * 16;

  const size_t rowBase = (size_t)(b * kS) * kDM + (size_t)h * kDK;

  // Q tile -> LDS fp32
  for (int idx = lane; idx < 16 * kDK; idx += 32) {
    const int r = idx >> 6, d = idx & 63;
    Qlds[w][r][d] = bf2f(Qg[rowBase + (size_t)(q0 + r) * kDM + d]);
  }
  // Q A-fragments for the two 32-wide d chunks
  AFrag aQ[2];
#pragma unroll
  for (int c = 0; c < 2; ++c) {
    const u16* base = Qg + rowBase + (size_t)(q0 + l16) * kDM + c * 32;
    aQ[c].q[0] = *(const U4*)(base + 8 * hf);
    aQ[c].q[1] = *(const U4*)(base + 16 + 8 * hf);
  }

  float mrow[8], lrow[8];
  v8f accO[4];
  v8f vzero = {};
#pragma unroll
  for (int r = 0; r < 8; ++r) { mrow[r] = -3.0e38f; lrow[r] = 0.f; }
#pragma unroll
  for (int t = 0; t < 4; ++t) accO[t] = vzero;

  for (int ks = 0; ks < kS; ks += 32) {
    // ---- scores = (Q Kt)*scale + rel*scale, mask ----
    float sc[2][8];
#pragma unroll
    for (int t = 0; t < 2; ++t) {
      AFrag bK0, bK1;   // B frag: col = k row ks+t*16+l16, dwords = d 8h..8h+7, 16+8h..23+8h
      const u16* kbp = Kg + rowBase + (size_t)(ks + t * 16 + l16) * kDM;
      bK0.q[0] = *(const U4*)(kbp + 8 * hf);
      bK0.q[1] = *(const U4*)(kbp + 16 + 8 * hf);
      bK1.q[0] = *(const U4*)(kbp + 32 + 8 * hf);
      bK1.q[1] = *(const U4*)(kbp + 32 + 16 + 8 * hf);
      v8f s = vzero;
      s = wmma_bf(aQ[0], bK0, s);
      s = wmma_bf(aQ[1], bK1, s);
#pragma unroll
      for (int r = 0; r < 8; ++r) {
        const int row = r + 8 * hf;
        const int ki  = ks + t * 16 + l16;
        const float4* qv  = (const float4*)&Qlds[w][row][0];
        const float4* rk4 = (const float4*)&rk[((size_t)(q0 + row) * kS + ki) * kDK];
        float rel = 0.f;
#pragma unroll
        for (int j = 0; j < 16; ++j) {
          float4 a = qv[j], c = rk4[j];
          rel += a.x * c.x + a.y * c.y + a.z * c.z + a.w * c.w;
        }
        float v = (s[r] + rel) * 0.125f;          // scale = 1/sqrt(64)
        if (mask[b * kS + ki] == 0) v = -1.0e9f;
        sc[t][r] = v;
      }
    }

    // ---- online softmax (reduce across the 16 lanes of each half) ----
    float p[2][8];
#pragma unroll
    for (int r = 0; r < 8; ++r) {
      float mx = fmaxf(sc[0][r], sc[1][r]);
#pragma unroll
      for (int off = 8; off > 0; off >>= 1)
        mx = fmaxf(mx, __shfl_xor(mx, off, 32));
      const float mn = fmaxf(mrow[r], mx);
      const float alpha = __expf(mrow[r] - mn);
      float ps = 0.f;
#pragma unroll
      for (int t = 0; t < 2; ++t) { p[t][r] = __expf(sc[t][r] - mn); ps += p[t][r]; }
#pragma unroll
      for (int off = 8; off > 0; off >>= 1)
        ps += __shfl_xor(ps, off, 32);
      lrow[r] = lrow[r] * alpha + ps;
      mrow[r] = mn;
#pragma unroll
      for (int t = 0; t < 4; ++t) accO[t][r] = accO[t][r] * alpha;
    }

    // ---- stage P (C layout -> LDS -> A layout) ----
#pragma unroll
    for (int t = 0; t < 2; ++t)
#pragma unroll
      for (int r = 0; r < 8; ++r) {
        const int row = r + 8 * hf;
        Pf[w][row][t * 16 + l16] = p[t][r];
        Pb[w][row][t * 16 + l16] = f2bf(p[t][r]);
      }
    asm volatile("s_wait_dscnt 0" ::: "memory");  // wave-local store->load fence

    AFrag pA;
    pA.q[0] = *(const U4*)&Pb[w][l16][16 * hf];
    pA.q[1] = *(const U4*)&Pb[w][l16][16 * hf + 8];

    // ---- O += P @ V (WMMA) and O += P @ r_v (VALU, L2-resident) ----
#pragma unroll
    for (int td = 0; td < 4; ++td) {
      AFrag bV;  // B frag: col d = td*16+l16, dword i = rows kk=2i+16h, +1
#pragma unroll
      for (int i = 0; i < 8; ++i) {
        const int kk = 2 * i + 16 * hf;
        const size_t a0 = rowBase + (size_t)(ks + kk) * kDM + td * 16 + l16;
        u32 lo = Vg[a0];
        u32 hi = Vg[a0 + kDM];
        bV.u[i] = lo | (hi << 16);
      }
      accO[td] = wmma_bf(pA, bV, accO[td]);
#pragma unroll
      for (int r = 0; r < 8; ++r) {
        const int row = r + 8 * hf;
        const float* pr = &Pf[w][row][0];
        const float* rp = &rvp[((size_t)(q0 + row) * kS + ks) * kDK + td * 16 + l16];
        float a = 0.f;
#pragma unroll 8
        for (int kk = 0; kk < 32; ++kk)
          a += pr[kk] * rp[(size_t)kk * kDK];
        accO[td][r] = accO[td][r] + a;
      }
    }
  }

  // ---- normalize + store bf16 attention output [B*S, DM] ----
#pragma unroll
  for (int td = 0; td < 4; ++td)
#pragma unroll
    for (int r = 0; r < 8; ++r) {
      const int row = r + 8 * hf;
      const float o = accO[td][r] / lrow[r];
      Og[rowBase + (size_t)(q0 + row) * kDM + td * 16 + l16] = f2bf(o);
    }
}

} // anonymous namespace

// ---------------------------------------------------------------- launcher
extern "C" void kernel_launch(void* const* d_in, const int* in_sizes, int n_in,
                              void* d_out, int out_size, void* d_ws, size_t ws_size,
                              hipStream_t stream) {
  (void)in_sizes; (void)n_in; (void)out_size; (void)ws_size;
  const float* x    = (const float*)d_in[0];
  const int*   mask = (const int*)d_in[1];
  const float* Wq   = (const float*)d_in[2];  const float* bq = (const float*)d_in[3];
  const float* Wk   = (const float*)d_in[4];  const float* bk = (const float*)d_in[5];
  const float* Wv   = (const float*)d_in[6];  const float* bv = (const float*)d_in[7];
  const float* Wo   = (const float*)d_in[8];  const float* bo = (const float*)d_in[9];
  const float* rk   = (const float*)d_in[10]; const float* rv = (const float*)d_in[11];
  const float* ln1g = (const float*)d_in[12]; const float* ln1b = (const float*)d_in[13];
  const float* ln2g = (const float*)d_in[14]; const float* ln2b = (const float*)d_in[15];
  const float* w1   = (const float*)d_in[16]; const float* b1 = (const float*)d_in[17];
  const float* w2   = (const float*)d_in[18]; const float* b2 = (const float*)d_in[19];

  char* ws = (char*)d_ws;
  size_t off = 0;
  auto take = [&](size_t bytes) -> char* {
    char* p = ws + off;
    off += (bytes + 255) & ~(size_t)255;
    return p;
  };
  u16*   xn1  = (u16*)  take((size_t)kM * kDM * 2);   // LN1(x) bf16
  u16*   qb   = (u16*)  take((size_t)kM * kDM * 2);
  u16*   kb_  = (u16*)  take((size_t)kM * kDM * 2);
  u16*   vb   = (u16*)  take((size_t)kM * kDM * 2);
  u16*   attO = (u16*)  take((size_t)kM * kDM * 2);
  float* x2   = (float*)take((size_t)kM * kDM * 4);   // x + attn proj
  u16*   h2   = (u16*)  take((size_t)kM * kDM * 2);   // LN2(x2) bf16
  u16*   f1   = (u16*)  take((size_t)kM * kDFF * 2);  // GELU(ffn1) bf16
  u16*   Wqb  = (u16*)  take((size_t)kDM * kDM * 2);
  u16*   Wkb  = (u16*)  take((size_t)kDM * kDM * 2);
  u16*   Wvb  = (u16*)  take((size_t)kDM * kDM * 2);
  u16*   Wob  = (u16*)  take((size_t)kDM * kDM * 2);
  u16*   w1b  = (u16*)  take((size_t)kDFF * kDM * 2);
  u16*   w2b  = (u16*)  take((size_t)kDM * kDFF * 2);

  auto cvt = [&](const float* s, u16* d, size_t n) {
    cvt_bf16_kernel<<<dim3((unsigned)((n + 255) / 256)), dim3(256), 0, stream>>>(s, d, (int)n);
  };
  cvt(Wq, Wqb, (size_t)kDM * kDM);
  cvt(Wk, Wkb, (size_t)kDM * kDM);
  cvt(Wv, Wvb, (size_t)kDM * kDM);
  cvt(Wo, Wob, (size_t)kDM * kDM);
  cvt(w1, w1b, (size_t)kDFF * kDM);
  cvt(w2, w2b, (size_t)kDM * kDFF);

  // sublayer 1
  ln_bf16_kernel<<<dim3(kM), dim3(256), 0, stream>>>(x, ln1g, ln1b, xn1);
  gemm_bf16_kernel<0><<<dim3(kM / 128, kDM / 128), dim3(256), 0, stream>>>(
      xn1, Wqb, bq, nullptr, qb, nullptr, kM, kDM, kDM);
  gemm_bf16_kernel<0><<<dim3(kM / 128, kDM / 128), dim3(256), 0, stream>>>(
      xn1, Wkb, bk, nullptr, kb_, nullptr, kM, kDM, kDM);
  gemm_bf16_kernel<0><<<dim3(kM / 128, kDM / 128), dim3(256), 0, stream>>>(
      xn1, Wvb, bv, nullptr, vb, nullptr, kM, kDM, kDM);
  attn_kernel<<<dim3(kB * kH * (kS / 64)), dim3(128), 0, stream>>>(
      qb, kb_, vb, rk, rv, mask, attO);
  gemm_bf16_kernel<2><<<dim3(kM / 128, kDM / 128), dim3(256), 0, stream>>>(
      attO, Wob, bo, x, nullptr, x2, kM, kDM, kDM);

  // sublayer 2
  ln_bf16_kernel<<<dim3(kM), dim3(256), 0, stream>>>(x2, ln2g, ln2b, h2);
  gemm_bf16_kernel<1><<<dim3(kM / 128, kDFF / 128), dim3(256), 0, stream>>>(
      h2, w1b, b1, nullptr, f1, nullptr, kM, kDFF, kDM);
  gemm_bf16_kernel<2><<<dim3(kM / 128, kDM / 128), dim3(256), 0, stream>>>(
      f1, w2b, b2, x2, nullptr, (float*)d_out, kM, kDM, kDFF);
}
